// MultiHeadAttention_67010079752239
// MI455X (gfx1250) — matrix operand order repaired and checked by reading, compiled, not run on hardware
//
#include <hip/hip_runtime.h>
#include <hip/hip_bf16.h>
#include <math.h>

// ---------------------------------------------------------------------------
// MI455X (gfx1250) multi-head attention forward.
//   1) cvt_weights:  w_q,w_k,w_v,w_o  f32 -> f16   (ws)
//   2) proj_kernel x3: Q/K/V projections (f32 in, f16 out)
//        Q,K -> [B,H,S,64]     V -> [B,H,64,S]  (transposed for PV B-frags)
//   3) flash_kernel: block-cooperative streaming attention.
//        K/V chunks staged in LDS once per block via the Tensor Data Mover
//        (tensor_load_to_lds, double buffered, s_wait_tensorcnt), online
//        softmax, all matmuls v_wmma_f32_16x16x32_f16.
//   4) proj_kernel: final O projection (f16 in, f32 out to d_out)
// ---------------------------------------------------------------------------

typedef __attribute__((ext_vector_type(16))) _Float16 v16h;
typedef __attribute__((ext_vector_type(8)))  _Float16 v8h;
typedef __attribute__((ext_vector_type(8)))  float    v8f;
typedef __attribute__((ext_vector_type(4)))  float    v4f;
typedef __attribute__((ext_vector_type(4)))  unsigned int u32x4;
typedef __attribute__((ext_vector_type(8)))  int      i32x8;
typedef __attribute__((ext_vector_type(4)))  int      i32x4;

#if __has_builtin(__builtin_amdgcn_tensor_load_to_lds) && __has_builtin(__builtin_amdgcn_s_wait_tensorcnt)
#define HAVE_TDM 1
#else
#define HAVE_TDM 0
#endif

__device__ __forceinline__ v8h ld8(const float* p) {
  v4f a = *(const v4f*)p;
  v4f b = *(const v4f*)(p + 4);
  v8h r = { (_Float16)a[0], (_Float16)a[1], (_Float16)a[2], (_Float16)a[3],
            (_Float16)b[0], (_Float16)b[1], (_Float16)b[2], (_Float16)b[3] };
  return r;
}
__device__ __forceinline__ v8h ld8(const _Float16* p) { return *(const v8h*)p; }

__device__ __forceinline__ v16h cat16(v8h lo, v8h hi) {
  return __builtin_shufflevector(lo, hi, 0,1,2,3,4,5,6,7,8,9,10,11,12,13,14,15);
}

// A-matrix fragment, 16x32 f16 (M x K), per ISA 7.12.2:
// lane l: row = l&15 ; K = kb..kb+7 and kb+16..kb+23, kb = (l>>4)*8
template<typename T>
__device__ __forceinline__ v16h load_a_frag(const T* base, int ld, int k0, int lane) {
  const T* p = base + (size_t)(lane & 15) * (size_t)ld + (size_t)(k0 + ((lane >> 4) << 3));
  return cat16(ld8(p), ld8(p + 16));
}

// B-matrix fragment, 32x16 f16 (K x N):
// lane l: col = l&15 ; K = kb..kb+7 and kb+16..kb+23, kb = (l>>4)*8
__device__ __forceinline__ v16h load_b_frag(const _Float16* base, int ld, int k0, int lane) {
  const _Float16* p = base + (size_t)(lane & 15) * (size_t)ld + (size_t)(k0 + ((lane >> 4) << 3));
  return cat16(*(const v8h*)p, *(const v8h*)(p + 16));
}

__device__ __forceinline__ v8f wmma16(v16h a, v16h b, v8f c) {
  return __builtin_amdgcn_wmma_f32_16x16x32_f16(false, a, false, b, (short)0, c, false, false);
}

__device__ __forceinline__ float rowmax16(float x) {
  x = fmaxf(x, __shfl_xor(x, 1, 16));
  x = fmaxf(x, __shfl_xor(x, 2, 16));
  x = fmaxf(x, __shfl_xor(x, 4, 16));
  x = fmaxf(x, __shfl_xor(x, 8, 16));
  return x;
}
__device__ __forceinline__ float rowsum16(float x) {
  x += __shfl_xor(x, 1, 16);
  x += __shfl_xor(x, 2, 16);
  x += __shfl_xor(x, 4, 16);
  x += __shfl_xor(x, 8, 16);
  return x;
}

#if HAVE_TDM
// LDS byte offset of a __shared__ object: generic shared address carries the
// workgroup-relative LDS offset in its low 32 bits (aperture in [63:32]).
__device__ __forceinline__ unsigned lds_off(const void* p) {
  return (unsigned)(uintptr_t)p;
}

// Issue a 2-D TDM tile load (D# per ISA 8.3-8.6). data_size = 2 bytes.
// tile (tile_d0 contiguous elems) x (tile_d1 rows), row stride = stride0 elems.
// LDS padding: (pad_amt_code+1) dwords every 2^(pad_int_code+1) dwords,
// to break LDS bank-conflict strides.
__device__ __forceinline__ void tdm_load_2d(unsigned ldsoff, unsigned long long gaddr,
                                            unsigned tensor_d0, unsigned tensor_d1,
                                            unsigned tile_d0, unsigned tile_d1,
                                            unsigned long long stride0,
                                            unsigned pad_int_code, unsigned pad_amt_code) {
  u32x4 g0;
  g0[0] = 1u;                                            // count=1 (valid), user mode
  g0[1] = ldsoff;                                        // lds_addr [63:32]
  g0[2] = (unsigned)(gaddr & 0xFFFFFFFFu);               // global_addr[31:0]
  g0[3] = (unsigned)((gaddr >> 32) & 0x1FFFFFFu)         // global_addr[56:32]
        | (2u << 30);                                    // type = 2 ("image")
  i32x8 g1;
  g1[0] = (int)((1u << 16)                               // data_size = 1 -> 2 bytes
              | (1u << 20)                               // pad_enable
              | (pad_int_code << 22)
              | (pad_amt_code << 25));
  g1[1] = (int)((tensor_d0 & 0xFFFFu) << 16);            // tensor_dim0[15:0] @ [63:48]
  g1[2] = (int)(((tensor_d0 >> 16) & 0xFFFFu)            // tensor_dim0[31:16]
              | ((tensor_d1 & 0xFFFFu) << 16));          // tensor_dim1[15:0]
  g1[3] = (int)(((tensor_d1 >> 16) & 0xFFFFu)            // tensor_dim1[31:16]
              | ((tile_d0 & 0xFFFFu) << 16));            // tile_dim0 @ [127:112]
  g1[4] = (int)(tile_d1 & 0xFFFFu);                      // tile_dim1; tile_dim2=0
  g1[5] = (int)(stride0 & 0xFFFFFFFFull);                // tensor_dim0_stride[31:0]
  g1[6] = (int)((stride0 >> 32) & 0xFFFFull);            // stride0[47:32]; stride1 lo=0
  g1[7] = 0;                                             // stride1 hi = 0
  i32x4 z4 = { 0, 0, 0, 0 };
  i32x8 z8 = { 0, 0, 0, 0, 0, 0, 0, 0 };
  // 6-arg form (clang-23 / therock headers): groups 0..3, extra group, cpol.
  __builtin_amdgcn_tensor_load_to_lds(g0, g1, z4, z4, z8, 0);
}
#endif

// ---------------------------------------------------------------------------
// Weight conversion: four 512x512 f32 matrices -> contiguous f16
// ---------------------------------------------------------------------------
__global__ __launch_bounds__(256)
void cvt_weights(const float* __restrict__ a, const float* __restrict__ b,
                 const float* __restrict__ c, const float* __restrict__ d,
                 _Float16* __restrict__ o, int n) {
  int i = blockIdx.x * 256 + threadIdx.x;
  if (i < n) {
    o[i]         = (_Float16)a[i];
    o[n + i]     = (_Float16)b[i];
    o[2 * n + i] = (_Float16)c[i];
    o[3 * n + i] = (_Float16)d[i];
  }
}

// ---------------------------------------------------------------------------
// Tiled GEMM:  out[m,n] = sum_k X[m,k] * W[n,k] + bias[n]   (torch Linear)
// One wave computes one 16x16 tile; 8 waves / block.
// m -> (batch, s) with 4096 rows/batch; n -> (h, d) with d_head = 64;
// offset = b*off_b + s*st_s + h*st_h + d*st_d.
// ---------------------------------------------------------------------------
template<typename InT, typename OutT>
__global__ __launch_bounds__(256)
void proj_kernel(const InT* __restrict__ X, const _Float16* __restrict__ W,
                 const float* __restrict__ bias, OutT* __restrict__ out,
                 int K, int ntiles,
                 long long off_b, long long st_s, long long st_h, long long st_d) {
  const int lane = threadIdx.x & 31;
  const int wave = threadIdx.x >> 5;
  const int tile = blockIdx.x * 8 + wave;
  const int tn = tile % ntiles;
  const int tm = tile / ntiles;

  const InT*      Xa = X + (size_t)tm * 16 * (size_t)K;
  const _Float16* Wb = W + (size_t)tn * 16 * (size_t)K;

  v8f acc = {};
  for (int kc = 0; kc < K; kc += 32) {
    v16h a = load_a_frag(Xa, K, kc, lane);
    v16h b = load_b_frag(Wb, K, kc, lane);
    acc = wmma16(a, b, acc);
  }

  const int n  = tn * 16 + (lane & 15);
  const float bv = bias[n];
  const int h = n >> 6;
  const int d = n & 63;
  const int mrow = tm * 16 + ((lane >> 4) << 3);
#pragma unroll
  for (int j = 0; j < 8; ++j) {
    int m  = mrow + j;
    int bb = m >> 12;        // 4096 rows per batch
    int s  = m & 4095;
    long long off = (long long)bb * off_b + (long long)s * st_s +
                    (long long)h * st_h + (long long)d * st_d;
    out[(size_t)off] = (OutT)(acc[j] + bv);
  }
}

// ---------------------------------------------------------------------------
// Flash attention. One block = 8 waves = 128 query rows of one (b,h).
// K chunk (32 keys x 64) and V chunk (64 x 32 keys, from transposed V) are
// staged in LDS once per block (TDM double-buffered when available), so each
// 32-key chunk is fetched 1x per 128 q-rows instead of 8x.
// LDS rows padded (+8 halves) to break 128B-stride bank conflicts; the TDM
// descriptor's pad_interval/pad_amount produce the same padded layout.
// ---------------------------------------------------------------------------
#define KPAD 72   // 64 + 8 halves per K row in LDS (144 B stride)
#define VPAD 40   // 32 + 8 halves per V row in LDS (80 B stride)

__global__ __launch_bounds__(256)
void flash_kernel(const _Float16* __restrict__ Qh, const _Float16* __restrict__ Kh,
                  const _Float16* __restrict__ Vt, _Float16* __restrict__ attn,
                  int S) {
#if HAVE_TDM
  __shared__ _Float16 ldsK[2][32 * KPAD];
  __shared__ _Float16 ldsV[2][64 * VPAD];
#else
  __shared__ _Float16 ldsK[1][32 * KPAD];
  __shared__ _Float16 ldsV[1][64 * VPAD];
#endif
  __shared__ _Float16 Plds[8][16 * 32];   // wave-private 16x32 P staging

  const int lane = threadIdx.x & 31;
  const int wave = threadIdx.x >> 5;
  const int tid  = threadIdx.x;
  const int bh   = blockIdx.y;                   // b*8 + h
  const int qbase = blockIdx.x * 128 + wave * 16;

  const _Float16* Qb = Qh + (size_t)bh * (size_t)S * 64;
  const _Float16* Kb = Kh + (size_t)bh * (size_t)S * 64;
  const _Float16* Vb = Vt + (size_t)bh * 64 * (size_t)S;

  // Preload the 16x64 Q tile as two A-fragments (K-chunks of 32).
  const v16h aq0 = load_a_frag(Qb + (size_t)qbase * 64, 64, 0,  lane);
  const v16h aq1 = load_a_frag(Qb + (size_t)qbase * 64, 64, 32, lane);

  float mrun[8], lrun[8];
  v8f acc0 = {}, acc1 = {}, acc2 = {}, acc3 = {};
#pragma unroll
  for (int j = 0; j < 8; ++j) { mrun[j] = -__builtin_inff(); lrun[j] = 0.0f; }

  const float sc = 0.125f;                       // 1/sqrt(d_head=64)
  _Float16* pw = &Plds[wave][0];

#if HAVE_TDM
  // Prologue: DMA chunk 0 into buffer 0 (wave 0 only; TDM is wave-level).
  if (wave == 0) {
    // K rows: 64 halves = 32 dwords -> pad_interval code 4; +4 dwords -> code 3
    tdm_load_2d(lds_off(&ldsK[0][0]), (unsigned long long)(uintptr_t)Kb,
                64u, (unsigned)S, 64u, 32u, 64ull, 4u, 3u);
    // V rows: 32 halves = 16 dwords -> pad_interval code 3; +4 dwords -> code 3
    tdm_load_2d(lds_off(&ldsV[0][0]), (unsigned long long)(uintptr_t)Vb,
                (unsigned)S, 64u, 32u, 64u, (unsigned long long)S, 3u, 3u);
  }
#endif

  for (int kc = 0; kc < S; kc += 32) {
#if HAVE_TDM
    const int buf = (kc >> 5) & 1;
    if (wave == 0) __builtin_amdgcn_s_wait_tensorcnt(0);
    __syncthreads();                              // chunk `buf` resident for all waves
    if (wave == 0 && (kc + 32) < S) {             // DMA next chunk into other buffer
      tdm_load_2d(lds_off(&ldsK[buf ^ 1][0]),
                  (unsigned long long)(uintptr_t)(Kb + (size_t)(kc + 32) * 64),
                  64u, (unsigned)S, 64u, 32u, 64ull, 4u, 3u);
      tdm_load_2d(lds_off(&ldsV[buf ^ 1][0]),
                  (unsigned long long)(uintptr_t)(Vb + (size_t)(kc + 32)),
                  (unsigned)S, 64u, 32u, 64u, (unsigned long long)S, 3u, 3u);
    }
    const _Float16* Kl = &ldsK[buf][0];
    const _Float16* Vl = &ldsV[buf][0];
#else
    // Cooperative synchronous staging (fallback): 256 threads x 16 B each.
    __syncthreads();                              // protect previous chunk's readers
    {
      int kr = tid >> 3, kcol = (tid & 7) * 8;    // K: 32 rows x 64 halves
      *(v8h*)&ldsK[0][kr * KPAD + kcol] = *(const v8h*)(Kb + (size_t)(kc + kr) * 64 + kcol);
      int vr = tid >> 2, vcol = (tid & 3) * 8;    // V: 64 rows x 32 halves
      *(v8h*)&ldsV[0][vr * VPAD + vcol] = *(const v8h*)(Vb + (size_t)vr * S + kc + vcol);
    }
    __syncthreads();
    const _Float16* Kl = &ldsK[0][0];
    const _Float16* Vl = &ldsV[0][0];
#endif

    // ---- scores: S = Q @ K^T, two 16x16 tiles covering 32 keys ----
    v8f s0 = {}, s1 = {};
    s0 = wmma16(aq0, load_b_frag(Kl,             KPAD, 0,  lane), s0);
    s0 = wmma16(aq1, load_b_frag(Kl,             KPAD, 32, lane), s0);
    s1 = wmma16(aq0, load_b_frag(Kl + 16 * KPAD, KPAD, 0,  lane), s1);
    s1 = wmma16(aq1, load_b_frag(Kl + 16 * KPAD, KPAD, 32, lane), s1);

    // ---- online softmax per query row (row j + 8*(lane>=16)) ----
#pragma unroll
    for (int j = 0; j < 8; ++j) {
      float x0 = s0[j] * sc;
      float x1 = s1[j] * sc;
      float mx = rowmax16(fmaxf(x0, x1));
      float mnew  = fmaxf(mrun[j], mx);
      float alpha = __expf(mrun[j] - mnew);
      float p0 = __expf(x0 - mnew);
      float p1 = __expf(x1 - mnew);
      lrun[j] = lrun[j] * alpha + rowsum16(p0 + p1);
      mrun[j] = mnew;
      acc0[j] *= alpha; acc1[j] *= alpha; acc2[j] *= alpha; acc3[j] *= alpha;
      // stage P (C-layout -> row-major 16x32 in LDS)
      int row = j + ((lane >> 4) << 3);
      pw[row * 32 + (lane & 15)]      = (_Float16)p0;
      pw[row * 32 + 16 + (lane & 15)] = (_Float16)p1;
    }

    // ---- reload P as A-fragment (16x32), accumulate O += P @ V ----
    v16h pa = load_a_frag((const _Float16*)pw, 32, 0, lane);
    acc0 = wmma16(pa, load_b_frag(Vl + 0  * VPAD, VPAD, 0, lane), acc0);
    acc1 = wmma16(pa, load_b_frag(Vl + 16 * VPAD, VPAD, 0, lane), acc1);
    acc2 = wmma16(pa, load_b_frag(Vl + 32 * VPAD, VPAD, 0, lane), acc2);
    acc3 = wmma16(pa, load_b_frag(Vl + 48 * VPAD, VPAD, 0, lane), acc3);
  }

  // ---- epilogue: normalize, write [B,S,512] with heads concatenated ----
  const int b = bh >> 3;
  const int h = bh & 7;
  const int half8 = (lane >> 4) << 3;
#pragma unroll
  for (int j = 0; j < 8; ++j) {
    float inv = 1.0f / lrun[j];
    int srow = qbase + j + half8;
    size_t base = ((size_t)b * (size_t)S + (size_t)srow) * 512 + (size_t)h * 64 + (lane & 15);
    attn[base + 0]  = (_Float16)(acc0[j] * inv);
    attn[base + 16] = (_Float16)(acc1[j] * inv);
    attn[base + 32] = (_Float16)(acc2[j] * inv);
    attn[base + 48] = (_Float16)(acc3[j] * inv);
  }
}

// ---------------------------------------------------------------------------
// Host launcher
// ---------------------------------------------------------------------------
extern "C" void kernel_launch(void* const* d_in, const int* in_sizes, int n_in,
                              void* d_out, int out_size, void* d_ws, size_t ws_size,
                              hipStream_t stream) {
  (void)in_sizes; (void)n_in; (void)out_size; (void)ws_size;
  const int B = 2, S = 4096, D = 512, H = 8, Dh = 64;
  const int M = B * S;                    // 8192 rows
  const int WE = D * D;                   // 262144 weight elements per matrix

  const float* q   = (const float*)d_in[0];
  const float* k   = (const float*)d_in[1];
  const float* v   = (const float*)d_in[2];
  const float* w_q = (const float*)d_in[3];
  const float* b_q = (const float*)d_in[4];
  const float* w_k = (const float*)d_in[5];
  const float* b_k = (const float*)d_in[6];
  const float* w_v = (const float*)d_in[7];
  const float* b_v = (const float*)d_in[8];
  const float* w_o = (const float*)d_in[9];
  const float* b_o = (const float*)d_in[10];

  // workspace layout (f16)
  char* ws = (char*)d_ws;
  _Float16* w16  = (_Float16*)ws;                          // 4 * 512 KB
  _Float16* wq16 = w16;
  _Float16* wk16 = w16 + WE;
  _Float16* wv16 = w16 + 2 * WE;
  _Float16* wo16 = w16 + 3 * WE;
  _Float16* Qh   = (_Float16*)(ws + (4ull * WE) * 2);                 // [B,H,S,64]  8 MB
  _Float16* Kh   = Qh + (size_t)B * H * S * Dh;                       // [B,H,S,64]  8 MB
  _Float16* Vt   = Kh + (size_t)B * H * S * Dh;                       // [B,H,64,S]  8 MB
  _Float16* At   = Vt + (size_t)B * H * S * Dh;                       // [B,S,512]   8 MB

  // 1) weights -> f16
  cvt_weights<<<(WE + 255) / 256, 256, 0, stream>>>(w_q, w_k, w_v, w_o, w16, WE);

  // 2) projections; grid = (M/16 * D/16) tiles / 8 waves per block
  const int ntiles = D / 16;                         // 32
  const int nblocks = (M / 16) * ntiles / 8;         // 2048
  const long long headBatch = (long long)H * S * Dh; // 2097152

  // Q,K -> [B,H,S,64]
  proj_kernel<float, _Float16><<<nblocks, 256, 0, stream>>>(
      q, wq16, b_q, Qh, D, ntiles, headBatch, Dh, (long long)S * Dh, 1);
  proj_kernel<float, _Float16><<<nblocks, 256, 0, stream>>>(
      k, wk16, b_k, Kh, D, ntiles, headBatch, Dh, (long long)S * Dh, 1);
  // V -> [B,H,64,S] (transposed)
  proj_kernel<float, _Float16><<<nblocks, 256, 0, stream>>>(
      v, wv16, b_v, Vt, D, ntiles, headBatch, 1, (long long)Dh * S, S);

  // 3) flash attention: one block per 128 q-rows per (b,h)
  dim3 fgrid(S / 128, B * H);                        // (32, 16)
  flash_kernel<<<fgrid, 256, 0, stream>>>(Qh, Kh, Vt, At, S);

  // 4) final projection -> f32 d_out [B,S,512]
  proj_kernel<_Float16, float><<<nblocks, 256, 0, stream>>>(
      At, wo16, b_o, (float*)d_out, D, ntiles,
      (long long)S * D, D, 64, 1);
}
